// UNetInnerProd_29506425324065
// MI455X (gfx1250) — matrix operand order repaired and checked by reading, not compiled, not run on hardware
//
#include <hip/hip_runtime.h>
#include <hip/hip_bf16.h>
#include <math.h>

typedef _Float16 v16h __attribute__((ext_vector_type(16)));
typedef _Float16 v8h  __attribute__((ext_vector_type(8)));
typedef float    v8f  __attribute__((ext_vector_type(8)));

#define NB    2
#define NC    16
#define NCP   32            // padded channel stride for f2h (K=16..31 stored as zeros)
#define NPIX  6912          // 72*96
#define NTI   432           // NPIX/16 tiles per axis
#define JCHK  27            // NTI/16 j-tile chunks
#define INV_DC 10.0f        // 1/DIST_COEF

// ---------------- workspace layout (bytes) ----------------
// [0,   64)            : 8 float scalars: per batch {S1, S2, fns, acc}
// [64,  +442368)       : f1h  : B*N*16 halfs (normalized+masked f1, f16)
// [..,  +884736)       : f2h  : B*N*32 halfs (normalized+masked f2, upper 16 = 0)
// [..,  +221184)       : p1   : B*N float4 {x,y,z,|p|^2}  (xyz1)
// [..,  +221184)       : p2t  : posed xyz2
// [..,  +221184)       : p2n  : xyz2
// total = 1,990,720 bytes

__global__ void init_scalars(float* scalars) {
  if (threadIdx.x < 8) scalars[threadIdx.x] = 0.0f;
}

__global__ __launch_bounds__(256)
void precompute_kernel(const float* __restrict__ f1, const float* __restrict__ f2,
                       const float* __restrict__ d1, const float* __restrict__ d2,
                       const float* __restrict__ pose, const float* __restrict__ yz1,
                       _Float16* __restrict__ f1h, _Float16* __restrict__ f2h,
                       float4* __restrict__ p1, float4* __restrict__ p2t,
                       float4* __restrict__ p2n, float* __restrict__ scalars) {
  int t = blockIdx.x * blockDim.x + threadIdx.x;
  if (t >= NB * NPIX) return;
  int b = t / NPIX;
  int n = t - b * NPIX;

  float dd1 = d1[t], dd2 = d2[t];
  float m1 = dd1 > 0.0f ? 1.0f : 0.0f;
  float m2 = dd2 > 0.0f ? 1.0f : 0.0f;

  float fa[NC], fb[NC];
  float sa = 0.0f, sb = 0.0f;
  #pragma unroll
  for (int c = 0; c < NC; ++c) {
    fa[c] = f1[((size_t)b * NC + c) * NPIX + n];
    fb[c] = f2[((size_t)b * NC + c) * NPIX + n];
    sa += fa[c] * fa[c];
    sb += fb[c] * fb[c];
  }
  float n1 = sqrtf(sa), n2 = sqrtf(sb);
  float sc1 = m1 / (n1 + 1e-8f);
  float sc2 = m2 / (n2 + 1e-8f);
  #pragma unroll
  for (int c = 0; c < NC; ++c) {
    f1h[(size_t)t * NC + c] = (_Float16)(fa[c] * sc1);
    f2h[(size_t)t * NCP + c] = (_Float16)(fb[c] * sc2);
    f2h[(size_t)t * NCP + NC + c] = (_Float16)0;   // K=16..31 zero padding
  }

  float gx = yz1[n], gy = yz1[NPIX + n], gz = yz1[2 * NPIX + n];
  float x1 = gx * dd1, y1 = gy * dd1, z1 = gz * dd1;
  p1[t] = make_float4(x1, y1, z1, x1 * x1 + y1 * y1 + z1 * z1);
  float x2 = gx * dd2, y2 = gy * dd2, z2 = gz * dd2;
  p2n[t] = make_float4(x2, y2, z2, x2 * x2 + y2 * y2 + z2 * z2);
  const float* P = pose + (size_t)b * 16;
  float tx = P[0] * x2 + P[1] * y2 + P[2]  * z2 + P[3];
  float ty = P[4] * x2 + P[5] * y2 + P[6]  * z2 + P[7];
  float tz = P[8] * x2 + P[9] * y2 + P[10] * z2 + P[11];
  p2t[t] = make_float4(tx, ty, tz, tx * tx + ty * ty + tz * tz);

  atomicAdd(&scalars[b * 4 + 0], m1);
  atomicAdd(&scalars[b * 4 + 1], m2);
  atomicAdd(&scalars[b * 4 + 2], n1 * m1 + n2 * m2);
}

// One wave per (batch, i-tile, chunk of 16 j-tiles). Gram tile via WMMA f16,
// RBF weights (two exps) in VALU, software-pipelined B / p2 loads (one iteration
// of latency hiding for every global load feeding the WMMA and the weight math).
__global__ __launch_bounds__(256)
void pair_reduce_kernel(const _Float16* __restrict__ f1h,
                        const _Float16* __restrict__ f2h,
                        const float4* __restrict__ p1,
                        const float4* __restrict__ p2t,
                        const float4* __restrict__ p2n,
                        float* __restrict__ scalars) {
  const int lane = threadIdx.x & 31;
  const int wave = threadIdx.x >> 5;
  const int gw   = blockIdx.x * (blockDim.x >> 5) + wave;

  const int perB = NTI * JCHK;
  const int b    = gw / perB;
  int rem        = gw - b * perB;
  const int ti   = rem / JCHK;
  const int jc   = rem - ti * JCHK;

  const int i0 = ti * 16;
  const int r  = lane & 15;   // column-in-tile for B/C; row-in-tile for A
  const int hi = lane >> 4;   // lane half selector

  // A tile (f1 normalized), row M = r, K = 16*hi + [0..15]; hi=1 lanes zero (K=16..31 pad).
  v16h a = {};
  if (hi == 0) {
    a = *(const v16h*)(f1h + ((size_t)(b * NPIX + i0 + r) * NC));
  }
  // Preload this lane's 8 C-tile rows (M = v + 8*hi) of xyz1 once.
  float4 q1[8];
  #pragma unroll
  for (int v = 0; v < 8; ++v) q1[v] = p1[b * NPIX + i0 + v + 8 * hi];

  float acc[8];
  #pragma unroll
  for (int v = 0; v < 8; ++v) acc[v] = 0.0f;

  // ---- software pipeline: prime iteration 0's loads ----
  const int jbase = jc * 16 * 16 + r;          // j for t=0
  v16h   bcur = *(const v16h*)(f2h + (size_t)(b * NPIX + jbase) * NCP + hi * NC);
  float4 qtc  = p2t[b * NPIX + jbase];
  float4 qnc  = p2n[b * NPIX + jbase];

  for (int t = 0; t < 16; ++t) {
    // prefetch t+1 (clamped on the last iteration; uniform, no divergence)
    const int tn    = (t < 15) ? (t + 1) : 15;
    const int jnext = (jc * 16 + tn) * 16 + r;
    v16h   bnx = *(const v16h*)(f2h + (size_t)(b * NPIX + jnext) * NCP + hi * NC);
    float4 qtn = p2t[b * NPIX + jnext];
    float4 qnn = p2n[b * NPIX + jnext];

    v8f c = {};
    c = __builtin_amdgcn_wmma_f32_16x16x32_f16(
        /*neg_a=*/false, a, /*neg_b=*/false, bcur,
        /*c_mod=*/(short)0, c, /*reuse_a=*/false, /*reuse_b=*/false);

    #pragma unroll
    for (int v = 0; v < 8; ++v) {
      float st = q1[v].w + qtc.w - 2.0f * (q1[v].x * qtc.x + q1[v].y * qtc.y + q1[v].z * qtc.z);
      float sn = q1[v].w + qnc.w - 2.0f * (q1[v].x * qnc.x + q1[v].y * qnc.y + q1[v].z * qnc.z);
      float kt = __expf(-fmaxf(st, 0.0f) * INV_DC);
      float kn = __expf(-fmaxf(sn, 0.0f) * INV_DC);
      acc[v] += (kt - kn) * c[v];
    }

    bcur = bnx;
    qtc  = qtn;
    qnc  = qnn;
  }

  // fold the 8 independent accumulators (tree), then wave32 butterfly reduction.
  float partial = ((acc[0] + acc[1]) + (acc[2] + acc[3]))
                + ((acc[4] + acc[5]) + (acc[6] + acc[7]));
  #pragma unroll
  for (int o = 16; o > 0; o >>= 1) partial += __shfl_xor(partial, o, 32);
  if (lane == 0) atomicAdd(&scalars[b * 4 + 3], partial);
}

__global__ void finalize_kernel(const float* __restrict__ scalars, float* __restrict__ out) {
  if (threadIdx.x != 0 || blockIdx.x != 0) return;
  float ineg = 0.0f, fns = 0.0f;
  #pragma unroll
  for (int b = 0; b < NB; ++b) {
    ineg += -scalars[b * 4 + 3] / (scalars[b * 4 + 0] * scalars[b * 4 + 1]);
    fns  += 100.0f * scalars[b * 4 + 2];
  }
  out[0] = ineg;   // final_loss (== inner_neg, sparse_mode=False)
  out[1] = ineg;   // inner_neg
  out[2] = fns;    // fea_norm_sum
}

extern "C" void kernel_launch(void* const* d_in, const int* in_sizes, int n_in,
                              void* d_out, int out_size, void* d_ws, size_t ws_size,
                              hipStream_t stream) {
  const float* f1   = (const float*)d_in[0];  // [B,C,H,W]
  const float* f2   = (const float*)d_in[1];
  const float* d1   = (const float*)d_in[2];  // [B,1,H,W]
  const float* d2   = (const float*)d_in[3];
  const float* pose = (const float*)d_in[4];  // [B,4,4]
  // d_in[5], d_in[6] (img1, img2) unused by the loss
  const float* yz1  = (const float*)d_in[7];  // [3,N]

  char* ws = (char*)d_ws;
  float*    scalars = (float*)ws;
  size_t o = 64;
  _Float16* f1h = (_Float16*)(ws + o); o += (size_t)NB * NPIX * NC  * sizeof(_Float16);
  _Float16* f2h = (_Float16*)(ws + o); o += (size_t)NB * NPIX * NCP * sizeof(_Float16);
  float4*   p1  = (float4*)(ws + o);   o += (size_t)NB * NPIX * sizeof(float4);
  float4*   p2t = (float4*)(ws + o);   o += (size_t)NB * NPIX * sizeof(float4);
  float4*   p2n = (float4*)(ws + o);

  init_scalars<<<1, 32, 0, stream>>>(scalars);

  {
    int nt = NB * NPIX;
    precompute_kernel<<<(nt + 255) / 256, 256, 0, stream>>>(
        f1, f2, d1, d2, pose, yz1, f1h, f2h, p1, p2t, p2n, scalars);
  }

  {
    // 2 batches * 432 i-tiles * 27 j-chunks = 23328 waves; 8 waves/block -> 2916 blocks, no tail.
    int waves  = NB * NTI * JCHK;
    int blocks = waves / 8;
    pair_reduce_kernel<<<blocks, 256, 0, stream>>>(f1h, f2h, p1, p2t, p2n, scalars);
  }

  finalize_kernel<<<1, 32, 0, stream>>>(scalars, (float*)d_out);
}
